// GNNModel_10599979286829
// MI455X (gfx1250) — compile-verified, matrix operand order checked by reading, not run on hardware
//
#include <hip/hip_runtime.h>
#include <hip/hip_bf16.h>
#include <math.h>

#define N_NODES 50000
#define C_IN    512
#define C_HID   128
#define E_EDGES 800000
#define EP      (E_EDGES + N_NODES)   // edges + self loops
#define HEADS2  3
#define C_OUT   2
#define NEG_SLOPE 0.2f

typedef __attribute__((ext_vector_type(16))) __bf16 v16bf;
typedef __attribute__((ext_vector_type(8)))  float  v8f;

union BfPack {
    unsigned short u[16];
    v16bf b;
};

// float -> bf16 bits, round-to-nearest-even
static __device__ inline unsigned short f2bf(float f) {
    unsigned int x = __float_as_uint(f);
    unsigned int r = (x + 0x7FFFu + ((x >> 16) & 1u)) >> 16;
    return (unsigned short)r;
}

// -------- generic fill --------
__global__ void fill_kernel(float* __restrict__ p, float v, int count) {
    int i = blockIdx.x * blockDim.x + threadIdx.x;
    if (i < count) p[i] = v;
}

// -------- GEMM1: h[N,128] = x[N,512] @ W1[512,128], bf16 WMMA, fp32 acc --------
// grid = N/16 row-blocks, 8 waves per block = one 16-col tile per wave.
__global__ void gemm1_kernel(const float* __restrict__ x,
                             const float* __restrict__ W1,
                             float* __restrict__ h) {
    const int m0   = blockIdx.x * 16;
    const int wave = threadIdx.x >> 5;     // 0..7 -> col tile
    const int lane = threadIdx.x & 31;
    const int n0   = wave * 16;
    const int lrow = lane & 15;
    const int khalf = (lane < 16) ? 0 : 8; // K sub-block per half-wave (ISA 7.12.2)

    v8f acc = {};
    for (int k0 = 0; k0 < C_IN; k0 += 32) {
        BfPack a, b;
        const float* arow = x  + (size_t)(m0 + lrow) * C_IN + (k0 + khalf);
        const float* bcol = W1 + (size_t)(k0 + khalf) * C_HID + (n0 + lrow);
#pragma unroll
        for (int j = 0; j < 8; ++j) {
            a.u[j]     = f2bf(arow[j]);                        // K = khalf + j
            a.u[j + 8] = f2bf(arow[16 + j]);                   // K = khalf + 16 + j
            b.u[j]     = f2bf(bcol[(size_t)j * C_HID]);        // row K = k0+khalf+j
            b.u[j + 8] = f2bf(bcol[(size_t)(16 + j) * C_HID]); // row K = +16
        }
        acc = __builtin_amdgcn_wmma_f32_16x16x32_bf16(
            false, a.b, false, b.b, (short)0, acc, false, false);
    }
    // C/D layout: VGPR v -> M = v (+8 for lanes 16..31), N = lane&15
    const int mo = (lane < 16) ? 0 : 8;
    float* out = h + (size_t)(m0 + mo) * C_HID + (n0 + lrow);
#pragma unroll
    for (int v = 0; v < 8; ++v) out[(size_t)v * C_HID] = acc[v];
}

// -------- per-node attention coefficients, layer 1 --------
__global__ void attn1_kernel(const float* __restrict__ h,
                             const float* __restrict__ att_src,
                             const float* __restrict__ att_dst,
                             float* __restrict__ as1, float* __restrict__ ad1) {
    int n = blockIdx.x * blockDim.x + threadIdx.x;
    if (n >= N_NODES) return;
    const float* hp = h + (size_t)n * C_HID;
    float s = 0.f, d = 0.f;
    for (int c = 0; c < C_HID; ++c) {
        float v = hp[c];
        s += v * att_src[c];
        d += v * att_dst[c];
    }
    as1[n] = s;
    ad1[n] = d;
}

static __device__ inline void edge_nodes(int i, const int* __restrict__ ei,
                                         int& s, int& d) {
    if (i < E_EDGES) { s = ei[i]; d = ei[E_EDGES + i]; }
    else             { s = d = i - E_EDGES; }            // self loop
}

// float atomic max via signed/unsigned int ordering (init = -inf)
static __device__ inline void atomicMaxF(float* addr, float val) {
    if (val >= 0.f) atomicMax((int*)addr, __float_as_int(val));
    else            atomicMin((unsigned int*)addr, (unsigned int)__float_as_int(val));
}

// -------- layer-1 edge passes --------
__global__ void edge_max1(const int* __restrict__ ei,
                          const float* __restrict__ as1, const float* __restrict__ ad1,
                          float* __restrict__ eb, float* __restrict__ m) {
    int i = blockIdx.x * blockDim.x + threadIdx.x;
    if (i >= EP) return;
    int s, d; edge_nodes(i, ei, s, d);
    float e = as1[s] + ad1[d];
    e = (e > 0.f) ? e : NEG_SLOPE * e;
    eb[i] = e;
    atomicMaxF(&m[d], e);
}

__global__ void edge_expsum1(const int* __restrict__ ei,
                             float* __restrict__ eb, const float* __restrict__ m,
                             float* __restrict__ den) {
    int i = blockIdx.x * blockDim.x + threadIdx.x;
    if (i >= EP) return;
    int s, d; edge_nodes(i, ei, s, d);
    float ex = __expf(eb[i] - m[d]);
    eb[i] = ex;
    atomicAdd(&den[d], ex);
}

// one wave per edge: lane covers 4 of the 128 features (coalesced 512B per edge)
__global__ void edge_agg1(const int* __restrict__ ei,
                          const float* __restrict__ eb, const float* __restrict__ den,
                          const float* __restrict__ h, float* __restrict__ acc) {
    int i = blockIdx.x * 8 + (threadIdx.x >> 5);
    if (i >= EP) return;
    int lane = threadIdx.x & 31;
    int s, d; edge_nodes(i, ei, s, d);
    float alpha = eb[i] / (den[d] + 1e-16f);
    const float* hs = h + (size_t)s * C_HID;
    float* od = acc + (size_t)d * C_HID;
#pragma unroll
    for (int j = 0; j < 4; ++j) {
        int c = lane + 32 * j;
        atomicAdd(&od[c], hs[c] * alpha);
    }
}

__global__ void relu_bias1(float* __restrict__ h1, const float* __restrict__ bias1) {
    int i = blockIdx.x * blockDim.x + threadIdx.x;
    if (i >= N_NODES * C_HID) return;
    float v = h1[i] + bias1[i & (C_HID - 1)];
    h1[i] = (v > 0.f) ? v : 0.f;
}

// -------- GEMM2: g[N,6] = h1[N,128] @ W2[128,6], one wave per 16-row block --------
__global__ void gemm2_kernel(const float* __restrict__ h1,
                             const float* __restrict__ W2,
                             float* __restrict__ g) {
    const int m0   = blockIdx.x * 16;
    const int lane = threadIdx.x & 31;
    const int lrow = lane & 15;
    const int khalf = (lane < 16) ? 0 : 8;
    v8f acc = {};
    for (int k0 = 0; k0 < C_HID; k0 += 32) {
        BfPack a, b;
        const float* arow = h1 + (size_t)(m0 + lrow) * C_HID + (k0 + khalf);
#pragma unroll
        for (int j = 0; j < 8; ++j) {
            a.u[j]     = f2bf(arow[j]);
            a.u[j + 8] = f2bf(arow[16 + j]);
            int kk0 = k0 + khalf + j;
            int kk1 = kk0 + 16;
            float b0 = (lrow < 6) ? W2[kk0 * 6 + lrow] : 0.0f;   // zero-pad N 6->16
            float b1 = (lrow < 6) ? W2[kk1 * 6 + lrow] : 0.0f;
            b.u[j]     = f2bf(b0);
            b.u[j + 8] = f2bf(b1);
        }
        acc = __builtin_amdgcn_wmma_f32_16x16x32_bf16(
            false, a.b, false, b.b, (short)0, acc, false, false);
    }
    const int mo = (lane < 16) ? 0 : 8;
    if (lrow < 6) {
        float* out = g + (size_t)(m0 + mo) * 6 + lrow;
#pragma unroll
        for (int v = 0; v < 8; ++v) out[(size_t)v * 6] = acc[v];
    }
}

// -------- per-node attention coefficients, layer 2 (3 heads x 2 ch) --------
__global__ void attn2_kernel(const float* __restrict__ g,
                             const float* __restrict__ att_src2,
                             const float* __restrict__ att_dst2,
                             float* __restrict__ as2, float* __restrict__ ad2) {
    int n = blockIdx.x * blockDim.x + threadIdx.x;
    if (n >= N_NODES) return;
    const float* gp = g + (size_t)n * 6;
#pragma unroll
    for (int hh = 0; hh < HEADS2; ++hh) {
        float g0 = gp[hh * 2], g1 = gp[hh * 2 + 1];
        as2[n * 3 + hh] = g0 * att_src2[hh * 2] + g1 * att_src2[hh * 2 + 1];
        ad2[n * 3 + hh] = g0 * att_dst2[hh * 2] + g1 * att_dst2[hh * 2 + 1];
    }
}

// -------- layer-2 edge passes (3 heads) --------
__global__ void edge_max2(const int* __restrict__ ei,
                          const float* __restrict__ as2, const float* __restrict__ ad2,
                          float* __restrict__ eb, float* __restrict__ m) {
    int i = blockIdx.x * blockDim.x + threadIdx.x;
    if (i >= EP) return;
    int s, d; edge_nodes(i, ei, s, d);
#pragma unroll
    for (int hh = 0; hh < HEADS2; ++hh) {
        float e = as2[s * 3 + hh] + ad2[d * 3 + hh];
        e = (e > 0.f) ? e : NEG_SLOPE * e;
        eb[i * 3 + hh] = e;
        atomicMaxF(&m[d * 3 + hh], e);
    }
}

__global__ void edge_expsum2(const int* __restrict__ ei,
                             float* __restrict__ eb, const float* __restrict__ m,
                             float* __restrict__ den) {
    int i = blockIdx.x * blockDim.x + threadIdx.x;
    if (i >= EP) return;
    int s, d; edge_nodes(i, ei, s, d);
#pragma unroll
    for (int hh = 0; hh < HEADS2; ++hh) {
        float ex = __expf(eb[i * 3 + hh] - m[d * 3 + hh]);
        eb[i * 3 + hh] = ex;
        atomicAdd(&den[d * 3 + hh], ex);
    }
}

__global__ void edge_agg2(const int* __restrict__ ei,
                          const float* __restrict__ eb, const float* __restrict__ den,
                          const float* __restrict__ g, float* __restrict__ o2) {
    int i = blockIdx.x * blockDim.x + threadIdx.x;
    if (i >= EP) return;
    int s, d; edge_nodes(i, ei, s, d);
#pragma unroll
    for (int hh = 0; hh < HEADS2; ++hh) {
        float alpha = eb[i * 3 + hh] / (den[d * 3 + hh] + 1e-16f);
        atomicAdd(&o2[d * 6 + hh * 2 + 0], g[s * 6 + hh * 2 + 0] * alpha);
        atomicAdd(&o2[d * 6 + hh * 2 + 1], g[s * 6 + hh * 2 + 1] * alpha);
    }
}

// -------- final: mean over heads + bias --------
__global__ void final_kernel(const float* __restrict__ o2,
                             const float* __restrict__ bias2,
                             float* __restrict__ out) {
    int i = blockIdx.x * blockDim.x + threadIdx.x;
    if (i >= N_NODES * C_OUT) return;
    int n = i >> 1, c = i & 1;
    float v = (o2[n * 6 + c] + o2[n * 6 + 2 + c] + o2[n * 6 + 4 + c]) * (1.0f / 3.0f);
    out[i] = v + bias2[c];
}

extern "C" void kernel_launch(void* const* d_in, const int* in_sizes, int n_in,
                              void* d_out, int out_size, void* d_ws, size_t ws_size,
                              hipStream_t stream) {
    const float* x        = (const float*)d_in[0];
    const int*   ei       = (const int*)  d_in[1];
    const float* W1       = (const float*)d_in[2];
    const float* att_src1 = (const float*)d_in[3];
    const float* att_dst1 = (const float*)d_in[4];
    const float* bias1    = (const float*)d_in[5];
    const float* W2       = (const float*)d_in[6];
    const float* att_src2 = (const float*)d_in[7];
    const float* att_dst2 = (const float*)d_in[8];
    const float* bias2    = (const float*)d_in[9];
    float* out = (float*)d_out;

    // workspace layout (fp32 elements)
    float* ws = (float*)d_ws;
    size_t o = 0;
    float* h    = ws + o; o += (size_t)N_NODES * C_HID;  // layer-1 projection
    float* h1   = ws + o; o += (size_t)N_NODES * C_HID;  // aggregated + relu
    float* as1  = ws + o; o += N_NODES;
    float* ad1  = ws + o; o += N_NODES;
    float* m1   = ws + o; o += N_NODES;
    float* den1 = ws + o; o += N_NODES;
    float* eb1  = ws + o; o += EP;
    float* g    = ws + o; o += (size_t)N_NODES * 6;
    float* as2  = ws + o; o += (size_t)N_NODES * 3;
    float* ad2  = ws + o; o += (size_t)N_NODES * 3;
    float* m2   = ws + o; o += (size_t)N_NODES * 3;
    float* den2 = ws + o; o += (size_t)N_NODES * 3;
    float* eb2  = ws + o; o += (size_t)EP * 3;
    float* o2   = ws + o; o += (size_t)N_NODES * 6;
    (void)ws_size; (void)n_in; (void)in_sizes; (void)out_size;

    const int T = 256;
    const float NEG_INF = -INFINITY;

    // ---- layer 1 ----
    gemm1_kernel<<<N_NODES / 16, 256, 0, stream>>>(x, W1, h);
    attn1_kernel<<<(N_NODES + T - 1) / T, T, 0, stream>>>(h, att_src1, att_dst1, as1, ad1);
    fill_kernel<<<(N_NODES + T - 1) / T, T, 0, stream>>>(m1, NEG_INF, N_NODES);
    fill_kernel<<<(N_NODES + T - 1) / T, T, 0, stream>>>(den1, 0.f, N_NODES);
    fill_kernel<<<((N_NODES * C_HID) + T - 1) / T, T, 0, stream>>>(h1, 0.f, N_NODES * C_HID);
    edge_max1<<<(EP + T - 1) / T, T, 0, stream>>>(ei, as1, ad1, eb1, m1);
    edge_expsum1<<<(EP + T - 1) / T, T, 0, stream>>>(ei, eb1, m1, den1);
    edge_agg1<<<(EP + 7) / 8, 256, 0, stream>>>(ei, eb1, den1, h, h1);
    relu_bias1<<<((N_NODES * C_HID) + T - 1) / T, T, 0, stream>>>(h1, bias1);

    // ---- layer 2 ----
    gemm2_kernel<<<N_NODES / 16, 32, 0, stream>>>(h1, W2, g);
    attn2_kernel<<<(N_NODES + T - 1) / T, T, 0, stream>>>(g, att_src2, att_dst2, as2, ad2);
    fill_kernel<<<(N_NODES * 3 + T - 1) / T, T, 0, stream>>>(m2, NEG_INF, N_NODES * 3);
    fill_kernel<<<(N_NODES * 3 + T - 1) / T, T, 0, stream>>>(den2, 0.f, N_NODES * 3);
    fill_kernel<<<(N_NODES * 6 + T - 1) / T, T, 0, stream>>>(o2, 0.f, N_NODES * 6);
    edge_max2<<<(EP + T - 1) / T, T, 0, stream>>>(ei, as2, ad2, eb2, m2);
    edge_expsum2<<<(EP + T - 1) / T, T, 0, stream>>>(ei, eb2, m2, den2);
    edge_agg2<<<(EP + T - 1) / T, T, 0, stream>>>(ei, eb2, den2, g, o2);
    final_kernel<<<((N_NODES * C_OUT) + T - 1) / T, T, 0, stream>>>(o2, bias2, out);
}
